// TorchAttention_17282948399753
// MI455X (gfx1250) — matrix operand order repaired and checked by reading, not compiled, not run on hardware
//
#include <hip/hip_runtime.h>

// ---------------------------------------------------------------------------
// CDNA5 (gfx1250) attention layer: QKV proj -> RoPE -> causal flash-attn -> out proj
// All matmuls via v_wmma_f32_16x16x32_bf16 (fp32 accumulate, bf16 operands).
// Global->LDS staging is software-pipelined: tile k+1's global loads are issued
// before tile k's WMMA compute so HBM latency hides under the tensor ops.
// ---------------------------------------------------------------------------

typedef __bf16 bf16_t;
typedef __attribute__((ext_vector_type(16))) __bf16 v16bf;
typedef __attribute__((ext_vector_type(8)))  __bf16 v8bf;
typedef __attribute__((ext_vector_type(8)))  float  v8f;

#define SEQ      2048
#define DIM      4096
#define N_HEADS  32
#define N_KV     8
#define HEAD_DIM 128
#define KV_DIM   (N_KV * HEAD_DIM)   // 1024

// Pack two 16-byte LDS chunks into one 16-element bf16 fragment register set.
__device__ __forceinline__ v16bf pack16(const bf16_t* lo, const bf16_t* hi) {
  v8bf a = *(const v8bf*)lo;
  v8bf b = *(const v8bf*)hi;
  v16bf r;
#pragma unroll
  for (int i = 0; i < 8; ++i) { r[i] = a[i]; r[8 + i] = b[i]; }
  return r;
}

__device__ __forceinline__ v8f wmma_bf16(v16bf a, v16bf b, v8f c) {
  return __builtin_amdgcn_wmma_f32_16x16x32_bf16(false, a, false, b, (short)0, c,
                                                 false, false);
}

__device__ __forceinline__ v8f zero_v8f() {
  v8f z = {0.f, 0.f, 0.f, 0.f, 0.f, 0.f, 0.f, 0.f};
  return z;
}

// ---------------------------------------------------------------------------
// Generic GEMM: C[M,N] = A[M,K] * B[K,N], fp32 in/out, bf16 WMMA inside.
// Block tile 128x128, K-step 64, 8 waves; wave tile 64x32 (4x2 WMMA tiles).
// Pipelined: global loads for tile i+1 are in flight during tile i compute.
// ---------------------------------------------------------------------------
#define GBM 128
#define GBN 128
#define GBK 64
#define GLD (GBK + 8)   // padded LDS row stride (bf16 elems); 144B, 16B aligned

__global__ __launch_bounds__(256) void gemm_bf16_wmma(
    const float* __restrict__ A, const float* __restrict__ B,
    float* __restrict__ C, int M, int N, int K) {
  __shared__ bf16_t As[GBM][GLD];   // row-major M x K tile
  __shared__ bf16_t Bt[GBN][GLD];   // transposed: N-major, K contiguous

  const int tid  = threadIdx.x;
  const int lane = tid & 31;
  const int wave = tid >> 5;
  const int half = lane & 15;
  const int hi   = lane >> 4;

  const int bm = blockIdx.y * GBM;
  const int bn = blockIdx.x * GBN;
  const int wm = (wave >> 2) * 64;   // 2 wave rows
  const int wn = (wave & 3) * 32;    // 4 wave cols

  // Staging thread mapping
  const int ac4 = tid & 15, ar0 = tid >> 4;   // A: 16 float4/row, 16 rows/pass
  const int bc4 = tid & 31, br0 = tid >> 5;   // B: 32 float4/row,  8 rows/pass

  v8f acc[4][2];
#pragma unroll
  for (int i = 0; i < 4; ++i)
#pragma unroll
    for (int j = 0; j < 2; ++j) acc[i][j] = zero_v8f();

  float4 ra[8], rb[8];   // in-flight global tile (registers)

  auto g_load = [&](int kb) {
#pragma unroll
    for (int rr = 0; rr < 8; ++rr)
      ra[rr] = *(const float4*)(A + (size_t)(bm + ar0 + rr * 16) * K + kb + ac4 * 4);
#pragma unroll
    for (int rr = 0; rr < 8; ++rr)
      rb[rr] = *(const float4*)(B + (size_t)(kb + br0 + rr * 8) * N + bn + bc4 * 4);
  };

  auto s_store = [&]() {
#pragma unroll
    for (int rr = 0; rr < 8; ++rr) {
      bf16_t* d = &As[ar0 + rr * 16][ac4 * 4];
      d[0] = (bf16_t)ra[rr].x; d[1] = (bf16_t)ra[rr].y;
      d[2] = (bf16_t)ra[rr].z; d[3] = (bf16_t)ra[rr].w;
    }
#pragma unroll
    for (int rr = 0; rr < 8; ++rr) {
      const int k = br0 + rr * 8, n = bc4 * 4;
      Bt[n + 0][k] = (bf16_t)rb[rr].x; Bt[n + 1][k] = (bf16_t)rb[rr].y;
      Bt[n + 2][k] = (bf16_t)rb[rr].z; Bt[n + 3][k] = (bf16_t)rb[rr].w;
    }
  };

  g_load(0);
  for (int kb = 0; kb < K; kb += GBK) {
    __syncthreads();          // previous compute done; LDS reusable
    s_store();                // registers -> bf16 LDS
    __syncthreads();          // tile visible to all waves
    if (kb + GBK < K) g_load(kb + GBK);   // prefetch next tile under compute

#pragma unroll
    for (int k0 = 0; k0 < GBK; k0 += 32) {
      v16bf bfr[2];
#pragma unroll
      for (int j = 0; j < 2; ++j) {
        const bf16_t* p = &Bt[wn + j * 16 + half][k0 + hi * 16];
        bfr[j] = pack16(p, p + 8);
      }
#pragma unroll
      for (int i = 0; i < 4; ++i) {
        const bf16_t* prow = &As[wm + i * 16 + half][0];
        v16bf afr = pack16(prow + k0 + hi * 16, prow + k0 + hi * 16 + 8);
        acc[i][0] = wmma_bf16(afr, bfr[0], acc[i][0]);
        acc[i][1] = wmma_bf16(afr, bfr[1], acc[i][1]);
      }
    }
  }

  // Epilogue: C/D layout -> VGPR g holds (M = g + 8*hi, N = half) of each tile.
#pragma unroll
  for (int i = 0; i < 4; ++i)
#pragma unroll
    for (int j = 0; j < 2; ++j)
#pragma unroll
      for (int g = 0; g < 8; ++g)
        C[(size_t)(bm + wm + i * 16 + g + hi * 8) * N + bn + wn + j * 16 + half] =
            acc[i][j][g];
}

// ---------------------------------------------------------------------------
// RoPE (interleaved pairs, matches reference reshape (..., d/2, 2)).
// buf: [S][H][128], freqs: [S][64][2] = (cos, sin)
// ---------------------------------------------------------------------------
__global__ void rope_kernel(float* __restrict__ buf, const float* __restrict__ freqs,
                            int S, int H) {
  const int idx   = blockIdx.x * blockDim.x + threadIdx.x;
  const int total = S * H * 64;
  if (idx >= total) return;
  const int p  = idx & 63;
  const int t  = idx >> 6;
  const int hh = t % H;
  const int s  = t / H;
  const float c = freqs[(s * 64 + p) * 2 + 0];
  const float d = freqs[(s * 64 + p) * 2 + 1];
  float* v = buf + ((size_t)s * H + hh) * HEAD_DIM + p * 2;
  const float a = v[0], b = v[1];
  v[0] = a * c - b * d;
  v[1] = a * d + b * c;
}

// ---------------------------------------------------------------------------
// Causal flash attention, GQA (4 Q-heads per KV head), start_pos = 0.
// Grid: (SEQ/128, N_HEADS). Block: 256 threads = 8 waves; wave owns 16 q rows.
// Q,O: [S][32][128] fp32.  K,V: [S][8][128] fp32.
// K/V tile staging pipelined like the GEMM.
// ---------------------------------------------------------------------------
__global__ __launch_bounds__(256) void attn_wmma_kernel(
    const float* __restrict__ Q, const float* __restrict__ Kb,
    const float* __restrict__ Vb, float* __restrict__ O) {
  __shared__ bf16_t Ks[32][HEAD_DIM + 8];   // [key][dim], padded (272B stride)
  __shared__ bf16_t Vt[HEAD_DIM][40];       // [dim][key], padded (80B stride)
  __shared__ bf16_t Pws[8][16][40];         // per-wave P scratch [row][key]

  const int h    = blockIdx.y;
  const int kvh  = h >> 2;                   // N_REP = 4
  const int qblk = blockIdx.x;
  const int tid  = threadIdx.x;
  const int lane = tid & 31;
  const int wave = tid >> 5;
  const int half = lane & 15;
  const int hi   = lane >> 4;
  const int qr0  = qblk * 128 + wave * 16;

  // Staging thread mapping: 32 float4 per 128-dim row, 8 keys per pass.
  const int c4 = tid & 31, r0 = tid >> 5;

  // Load per-wave Q fragments (A layout), scaled by 1/sqrt(128), as bf16.
  const float scale = 0.08838834764831845f;   // 128^-0.5
  v16bf qf[4];
  {
    const float* qrow = Q + (size_t)(qr0 + half) * (N_HEADS * HEAD_DIM) + h * HEAD_DIM;
#pragma unroll
    for (int ks = 0; ks < 4; ++ks) {
      const int kd = ks * 32;
      v16bf f;
#pragma unroll
      for (int i = 0; i < 8; ++i) {
        f[i]     = (bf16_t)(qrow[kd + hi * 16 + i] * scale);
        f[8 + i] = (bf16_t)(qrow[kd + hi * 16 + 8 + i] * scale);
      }
      qf[ks] = f;
    }
  }

  float mrun[8], lrun[8];
#pragma unroll
  for (int g = 0; g < 8; ++g) { mrun[g] = -1e30f; lrun[g] = 0.f; }
  v8f accO[8];
#pragma unroll
  for (int t = 0; t < 8; ++t) accO[t] = zero_v8f();

  float4 rk[4], rv[4];   // in-flight K/V tile (registers)

  auto kv_load = [&](int kv0) {
#pragma unroll
    for (int rr = 0; rr < 4; ++rr) {
      const size_t base = ((size_t)(kv0 + r0 + rr * 8) * N_KV + kvh) * HEAD_DIM + c4 * 4;
      rk[rr] = *(const float4*)(Kb + base);
      rv[rr] = *(const float4*)(Vb + base);
    }
  };

  auto kv_store = [&]() {
#pragma unroll
    for (int rr = 0; rr < 4; ++rr) {
      const int key = r0 + rr * 8;
      bf16_t* d = &Ks[key][c4 * 4];
      d[0] = (bf16_t)rk[rr].x; d[1] = (bf16_t)rk[rr].y;
      d[2] = (bf16_t)rk[rr].z; d[3] = (bf16_t)rk[rr].w;
      const int dd = c4 * 4;
      Vt[dd + 0][key] = (bf16_t)rv[rr].x; Vt[dd + 1][key] = (bf16_t)rv[rr].y;
      Vt[dd + 2][key] = (bf16_t)rv[rr].z; Vt[dd + 3][key] = (bf16_t)rv[rr].w;
    }
  };

  const int ntiles = (qblk + 1) * 4;   // 32-key tiles covering keys <= qblk*128+127
  kv_load(0);
  for (int t = 0; t < ntiles; ++t) {
    const int kv0 = t * 32;
    __syncthreads();          // previous compute done
    kv_store();               // registers -> bf16 LDS (K row-major, V transposed)
    __syncthreads();
    if (t + 1 < ntiles) kv_load((t + 1) * 32);   // prefetch under compute

    // S = Q * K^T for two 16x16 tiles (keys kv0..+15 and kv0+16..+31).
    v8f s0 = zero_v8f(), s1 = zero_v8f();
#pragma unroll
    for (int ks = 0; ks < 4; ++ks) {
      const int kd = ks * 32;
      const bf16_t* p0 = &Ks[half][kd + hi * 16];
      const bf16_t* p1 = &Ks[16 + half][kd + hi * 16];
      s0 = wmma_bf16(qf[ks], pack16(p0, p0 + 8), s0);
      s1 = wmma_bf16(qf[ks], pack16(p1, p1 + 8), s1);
    }

    // Causal mask + online softmax (row = g + 8*hi, col = half within tile).
#pragma unroll
    for (int g = 0; g < 8; ++g) {
      const int qidx = qr0 + g + hi * 8;
      float a = (kv0 + half      <= qidx) ? s0[g] : -1e30f;
      float b = (kv0 + 16 + half <= qidx) ? s1[g] : -1e30f;
      float mx = fmaxf(a, b);
#pragma unroll
      for (int x = 1; x < 16; x <<= 1) mx = fmaxf(mx, __shfl_xor(mx, x, 32));
      const float mnew  = fmaxf(mrun[g], mx);
      const float alpha = __expf(mrun[g] - mnew);
      const float e0 = __expf(a - mnew);
      const float e1 = __expf(b - mnew);
      float sum = e0 + e1;
#pragma unroll
      for (int x = 1; x < 16; x <<= 1) sum += __shfl_xor(sum, x, 32);
      lrun[g] = lrun[g] * alpha + sum;
      mrun[g] = mnew;
#pragma unroll
      for (int dt = 0; dt < 8; ++dt) accO[dt][g] *= alpha;
      // P tile -> per-wave LDS scratch (C layout -> memory [row][key]).
      Pws[wave][g + hi * 8][half]      = (bf16_t)e0;
      Pws[wave][g + hi * 8][16 + half] = (bf16_t)e1;
    }

    // O += P * V   (P: 16x32 A fragment; V: 32x16 B fragments per 16-dim slice)
    const bf16_t* prow = &Pws[wave][half][0];
    v16bf pf = pack16(prow + hi * 16, prow + hi * 16 + 8);
#pragma unroll
    for (int dt = 0; dt < 8; ++dt) {
      const bf16_t* vp = &Vt[dt * 16 + half][hi * 16];
      accO[dt] = wmma_bf16(pf, pack16(vp, vp + 8), accO[dt]);
    }
  }

  // Normalize and write O[s][h][d].
#pragma unroll
  for (int g = 0; g < 8; ++g) {
    const float invl = 1.f / lrun[g];
    const size_t row = (size_t)(qr0 + g + hi * 8) * (N_HEADS * HEAD_DIM) + h * HEAD_DIM;
#pragma unroll
    for (int dt = 0; dt < 8; ++dt)
      O[row + dt * 16 + half] = accO[dt][g] * invl;
  }
}

// ---------------------------------------------------------------------------
// Host-side launch
// Inputs: 0:x 1:cache_kv 2:freqs_cis 3:wq 4:wk 5:wv 6:wo 7:start_pos(=0)
// ---------------------------------------------------------------------------
extern "C" void kernel_launch(void* const* d_in, const int* in_sizes, int n_in,
                              void* d_out, int out_size, void* d_ws, size_t ws_size,
                              hipStream_t stream) {
  const float* x     = (const float*)d_in[0];
  const float* freqs = (const float*)d_in[2];
  const float* wq    = (const float*)d_in[3];
  const float* wk    = (const float*)d_in[4];
  const float* wv    = (const float*)d_in[5];
  const float* wo    = (const float*)d_in[6];
  float* out = (float*)d_out;

  char* ws = (char*)d_ws;
  float* q_buf = (float*)(ws);
  float* k_buf = (float*)(ws + (size_t)SEQ * DIM * 4);
  float* v_buf = (float*)(ws + (size_t)SEQ * DIM * 4 + (size_t)SEQ * KV_DIM * 4);
  float* a_buf = (float*)(ws + (size_t)SEQ * DIM * 4 + 2 * (size_t)SEQ * KV_DIM * 4);

  // 1) Projections (fp32 -> bf16 staged inside GEMM, fp32 accumulate).
  gemm_bf16_wmma<<<dim3(DIM / GBN, SEQ / GBM), 256, 0, stream>>>(x, wq, q_buf, SEQ, DIM, DIM);
  gemm_bf16_wmma<<<dim3(KV_DIM / GBN, SEQ / GBM), 256, 0, stream>>>(x, wk, k_buf, SEQ, KV_DIM, DIM);
  gemm_bf16_wmma<<<dim3(KV_DIM / GBN, SEQ / GBM), 256, 0, stream>>>(x, wv, v_buf, SEQ, KV_DIM, DIM);

  // 2) RoPE on Q and K.
  {
    const int nq = SEQ * N_HEADS * 64;
    rope_kernel<<<(nq + 255) / 256, 256, 0, stream>>>(q_buf, freqs, SEQ, N_HEADS);
    const int nk = SEQ * N_KV * 64;
    rope_kernel<<<(nk + 255) / 256, 256, 0, stream>>>(k_buf, freqs, SEQ, N_KV);
  }

  // 3) Causal flash attention (start_pos = 0).
  attn_wmma_kernel<<<dim3(SEQ / 128, N_HEADS), 256, 0, stream>>>(q_buf, k_buf, v_buf, a_buf);

  // 4) Output projection.
  gemm_bf16_wmma<<<dim3(DIM / GBN, SEQ / GBM), 256, 0, stream>>>(a_buf, wo, out, SEQ, DIM, DIM);
}
